// RecurrentLIFNode_56831007261292
// MI455X (gfx1250) — matrix operand order repaired and checked by reading, not compiled, not run on hardware
//
#include <hip/hip_runtime.h>
#include <hip/hip_bf16.h>

typedef __attribute__((ext_vector_type(16))) _Float16 v16h;
typedef __attribute__((ext_vector_type(8)))  float    v8f;

#define TDIM   128
#define BDIM   512
#define HDIM   512
#define KDIM   512
#define MDIM   (TDIM * BDIM)      // 65536 rows of x flattened
#define BH     ((size_t)(BDIM * HDIM))
#define NGRP   8                  // 512 / 64 column groups
#define MPAIRS (MDIM / 32)        // 2048 wave tasks (2 M-tiles each)

// ---------------------------------------------------------------------------
// GEMM: xp[m, n] = sum_k x[m,k] * W[n,k] + bias[n]
// Block = 8 waves, one 64-column N group. W slice staged once in LDS as f16
// in WMMA B-fragment layout (64 KB). Each wave computes two 16-row M tiles
// -> 8 WMMAs per K step against LDS-resident B fragments.
// ---------------------------------------------------------------------------
__global__ void __launch_bounds__(256)
lif_proj_gemm_f16wmma(const float* __restrict__ X,
                      const float* __restrict__ W,
                      const float* __restrict__ bias,
                      float* __restrict__ XP) {
    // [k0idx(16)][j(4)][lane(32)][16 halfs] = 64 KB
    __shared__ alignas(32) _Float16 ldsB[16 * 4 * 32 * 16];

    const int lane   = threadIdx.x & 31;
    const int wave   = threadIdx.x >> 5;      // 0..7
    const int ngroup = blockIdx.x & 7;
    const int n0     = ngroup * 64;
    const int row    = lane & 15;             // M row (A) / N col (B) in tile
    const int hi     = lane >> 4;             // 0: lanes 0-15, 1: lanes 16-31

    // ---- Stage W tile (64 cols x 512 K) as f16 fragments in LDS ----------
    // 64 fragments (k0idx, j); each wave stages 8. B-fragment layout:
    // lane holds column n0+j*16+row; lo lanes K=k0+0..7 & 16..23, hi lanes K=k0+8..15 & 24..31.
    for (int f = wave; f < 64; f += 8) {
        const int k0i = f >> 2;
        const int j   = f & 3;
        const int col = n0 + j * 16 + row;
        const float* bp = W + (size_t)col * KDIM + k0i * 32 + hi * 8;
        _Float16* dst = &ldsB[((size_t)f * 32 + lane) * 16];
#pragma unroll
        for (int e = 0; e < 16; ++e) dst[e] = (_Float16)bp[e < 8 ? e : e + 8];
    }
    __syncthreads();

    // ---- Two 16-row M tiles per wave against the 64-col N group ----------
    const int pairBase = ((blockIdx.x >> 3) << 3) + wave;   // 0..2047
    const int mA = pairBase * 32;
    const int mB = mA + 16;

    v8f acc[2][4];
#pragma unroll
    for (int t = 0; t < 2; ++t)
#pragma unroll
        for (int j = 0; j < 4; ++j) acc[t][j] = (v8f){};

    // A fragment: lane holds row m+row; lo lanes K+0..7 & K+16..23,
    // hi lanes K+8..15 & K+24..31 (16-bit A 16x32 VGPR layout).
    const float* aBase0 = X + (size_t)(mA + row) * KDIM + hi * 8;
    const float* aBase1 = X + (size_t)(mB + row) * KDIM + hi * 8;

    for (int k0i = 0; k0i < 16; ++k0i) {
        const float* ap0 = aBase0 + k0i * 32;
        const float* ap1 = aBase1 + k0i * 32;
        v16h a0, a1;
#pragma unroll
        for (int e = 0; e < 8; ++e) {
            a0[e]     = (_Float16)ap0[e];
            a0[8 + e] = (_Float16)ap0[16 + e];
            a1[e]     = (_Float16)ap1[e];
            a1[8 + e] = (_Float16)ap1[16 + e];
        }

#pragma unroll
        for (int j = 0; j < 4; ++j) {
            const v16h bfrag =
                *(const v16h*)&ldsB[(((size_t)k0i * 4 + j) * 32 + lane) * 16];
            acc[0][j] = __builtin_amdgcn_wmma_f32_16x16x32_f16(
                false, a0, false, bfrag, (short)0, acc[0][j], false, false);
            acc[1][j] = __builtin_amdgcn_wmma_f32_16x16x32_f16(
                false, a1, false, bfrag, (short)0, acc[1][j], false, false);
        }
    }

    // C/D layout: VGPR r -> (M = r + hi*8, N = lane&15).
#pragma unroll
    for (int t = 0; t < 2; ++t) {
        const int m0 = (t == 0) ? mA : mB;
#pragma unroll
        for (int j = 0; j < 4; ++j) {
            const int n    = n0 + j * 16 + row;
            const float bv = bias[n];
#pragma unroll
            for (int r = 0; r < 8; ++r) {
                const int m = m0 + r + hi * 8;
                XP[(size_t)m * HDIM + n] = acc[t][j][r] + bv;
            }
        }
    }
}

// ---------------------------------------------------------------------------
// Recurrent LIF scan: one thread per (b,h); 4 layer membrane states live in
// registers across all 128 timesteps. Reads xp[t] and overwrites the same
// address with the averaged last-layer spike (xp aliases out by design).
// ---------------------------------------------------------------------------
__global__ void __launch_bounds__(256)
lif_scan_kernel(const float* xp,          // aliases `out` — no __restrict__
                float* out,
                float* __restrict__ states_last) {
    const size_t idx = (size_t)blockIdx.x * blockDim.x + threadIdx.x; // 0..BH-1

    float v[4]  = {0.f, 0.f, 0.f, 0.f};
    float av[4] = {0.f, 0.f, 0.f, 0.f};

    for (int t = 0; t < TDIM; ++t) {
        float y = xp[(size_t)t * BH + idx];
        float last_s = 0.f;
#pragma unroll
        for (int l = 0; l < 4; ++l) {
            float vv    = v[l];
            float acc_v = 0.f;
            float acc_s = 0.f;
#pragma unroll
            for (int s = 0; s < 4; ++s) {
                vv = vv + (y - vv) * 0.5f;                    // charge, tau=2
                const float sp = (vv >= 1.0f) ? 1.0f : 0.0f;  // heaviside(v-Vth)
                vv -= sp;                                     // soft reset
                acc_v += vv;
                acc_s += sp;
            }
            v[l]  = vv;
            av[l] = acc_v * 0.25f;
            y      = av[l];                                   // next layer input
            last_s = acc_s * 0.25f;
        }
        out[(size_t)t * BH + idx] = last_s;

        if (t == TDIM - 1) {
#pragma unroll
            for (int l = 0; l < 4; ++l)
                states_last[(size_t)l * BH + idx] = av[l];
        }
    }
}

// ---------------------------------------------------------------------------
extern "C" void kernel_launch(void* const* d_in, const int* in_sizes, int n_in,
                              void* d_out, int out_size, void* d_ws, size_t ws_size,
                              hipStream_t stream) {
    (void)in_sizes; (void)n_in; (void)d_ws; (void)ws_size; (void)out_size;

    const float* x  = (const float*)d_in[0];  // [T, B, INPUT]
    const float* W  = (const float*)d_in[1];  // [H, INPUT]
    const float* bv = (const float*)d_in[2];  // [H]

    float* out    = (float*)d_out;                 // first T*B*H floats
    float* states = out + (size_t)TDIM * BH;       // next 4*B*H floats

    // Stage 1: projection GEMM into the `out` region (xp scratch; each element
    // is consumed exactly once by its owning scan thread, then overwritten).
    // Grid: 8 N-groups x 256 blocks of 8 wave-tasks (2 M-tiles each) = 2048.
    lif_proj_gemm_f16wmma<<<2048, 256, 0, stream>>>(x, W, bv, out);

    // Stage 2: full T-scan with register-resident layer states.
    lif_scan_kernel<<<(int)(BH / 256), 256, 0, stream>>>(out, out, states);
}
